// Retriever_44890998178258
// MI455X (gfx1250) — matrix operand order repaired and checked by reading, not compiled, not run on hardware
//
#include <hip/hip_runtime.h>

typedef __attribute__((ext_vector_type(16))) _Float16 v16h;
typedef __attribute__((ext_vector_type(8)))  float    v8f;

#define NUM_HEADS 4
#define OUT_DIM   256
#define BATCH     64
#define HID       512
#define NQ        256            // NUM_HEADS * BATCH query rows
#define D         256            // feature dim
#define NTOT      500000         // index size
#define TOPK      16
#define NWG_A     125            // workgroups in score kernel
#define TILES_WG  250            // NWG_A * TILES_WG * 16 == NTOT exactly
#define BROW      264            // padded LDS row stride (f16 elems) -> conflict-free b128 reads
#define NCAND     (NWG_A * TOPK) // 2000 candidates per query
#define CPL       63             // ceil(NCAND / 32) per lane

// ---------------------------------------------------------------------------
// Kernel 1: q = LayerNorm(reshape(hidden @ Wp + bp)) ; emit f16 hi/lo split.
// Row q = h*64 + b (matches [H,B,D] transpose+reshape in the reference).
// ---------------------------------------------------------------------------
__global__ __launch_bounds__(256) void rt_encode_q(
    const float* __restrict__ hidden, const float* __restrict__ Wp,
    const float* __restrict__ bp, _Float16* __restrict__ qhi,
    _Float16* __restrict__ qlo) {
  __shared__ float hrow[HID];
  __shared__ float red[256];
  __shared__ float red2[256];
  const int q = blockIdx.x;      // 0..255
  const int h = q >> 6;
  const int b = q & 63;
  const int j = threadIdx.x;     // 0..255
  hrow[j]       = hidden[b * HID + j];
  hrow[j + 256] = hidden[b * HID + 256 + j];
  __syncthreads();
  const int col = h * OUT_DIM + j;
  float acc = bp[col];
  #pragma unroll 8
  for (int k = 0; k < HID; ++k)
    acc = fmaf(hrow[k], Wp[k * (NUM_HEADS * OUT_DIM) + col], acc);
  red[j] = acc;
  red2[j] = acc * acc;
  __syncthreads();
  for (int s = 128; s > 0; s >>= 1) {
    if (j < s) { red[j] += red[j + s]; red2[j] += red2[j + s]; }
    __syncthreads();
  }
  const float mean = red[0] * (1.0f / 256.0f);
  const float var  = red2[0] * (1.0f / 256.0f) - mean * mean;
  const float y = (acc - mean) * rsqrtf(var + 1e-5f);
  const _Float16 hi = (_Float16)y;
  const _Float16 lo = (_Float16)(y - (float)hi);
  qhi[q * D + j] = hi;
  qlo[q * D + j] = lo;
}

// ---------------------------------------------------------------------------
// Kernel 2: streaming scores + exact per-chunk top-16.
// 512 threads = 16 waves; wave w owns query rows 16w..16w+15.
// Split-f16 WMMA: score = hi_q*hi_m + hi_q*lo_m + lo_q*hi_m (f32 accum).
// ---------------------------------------------------------------------------
__global__ __launch_bounds__(512) void rt_score_topk(
    const _Float16* __restrict__ qhi, const _Float16* __restrict__ qlo,
    const float* __restrict__ mem, float* __restrict__ candS,
    int* __restrict__ candC) {
  __shared__ alignas(16) _Float16 Bhi[16 * BROW];
  __shared__ alignas(16) _Float16 Blo[16 * BROW];
  __shared__ float topS[NQ * TOPK];
  __shared__ int   topC[NQ * TOPK];
  __shared__ float minV[NQ];

  const int tid   = threadIdx.x;
  const int wave  = tid >> 5;
  const int lane  = tid & 31;
  const int qbase = wave * 16;

  for (int i = tid; i < NQ * TOPK; i += 512) { topS[i] = -__builtin_inff(); topC[i] = 0; }
  for (int i = tid; i < NQ; i += 512) minV[i] = -__builtin_inff();

  // A operand (16x256, hi & lo) in WMMA register layout:
  // lane l holds row M=l%16; VGPR slots 0-7 hold K = k0 + (l<16?0:8) + {0..7},
  // slots 8-15 hold K = k0 + 16 + (l<16?0:8) + {0..7}  -> two b128 loads each.
  const int arow  = qbase + (lane & 15);
  const int akoff = (lane < 16) ? 0 : 8;
  v16h Ahi[8], Alo[8];
  #pragma unroll
  for (int kb = 0; kb < 8; ++kb) {
    const int c = kb * 32 + akoff;
    union { v16h v; uint4 u[2]; } t;
    t.u[0] = *(const uint4*)(qhi + arow * D + c);
    t.u[1] = *(const uint4*)(qhi + arow * D + c + 16);
    Ahi[kb] = t.v;
    t.u[0] = *(const uint4*)(qlo + arow * D + c);
    t.u[1] = *(const uint4*)(qlo + arow * D + c + 16);
    Alo[kb] = t.v;
  }
  __syncthreads();

  const int n_start = blockIdx.x * (TILES_WG * 16);
  const int nl = (tid * 8) >> 8;   // local column this thread fills (0..15)
  const int kf = (tid * 8) & 255;  // k offset this thread fills

  for (int t = 0; t < TILES_WG; ++t) {
    const int n0 = n_start + t * 16;
    // coalesced fp32 tile load (kept in regs until the barrier)
    const float* src = mem + (size_t)(n0 + nl) * D + kf;
    const float4 x0 = ((const float4*)src)[0];
    const float4 x1 = ((const float4*)src)[1];
    if (t + 1 < TILES_WG) __builtin_prefetch(src + 16 * D, 0, 1);

    __syncthreads();  // previous tile fully consumed
    {
      union { _Float16 h[8]; uint4 u; } ph, pl;
      const float xs[8] = {x0.x, x0.y, x0.z, x0.w, x1.x, x1.y, x1.z, x1.w};
      #pragma unroll
      for (int i = 0; i < 8; ++i) {
        const float x = xs[i];
        const _Float16 hh = (_Float16)x;
        ph.h[i] = hh;
        pl.h[i] = (_Float16)(x - (float)hh);
      }
      *(uint4*)(Bhi + nl * BROW + kf) = ph.u;
      *(uint4*)(Blo + nl * BROW + kf) = pl.u;
    }
    __syncthreads();  // tile ready

    // B operand layout: lane l holds column N=l%16; slots 0-7 hold K = (l<16?0:8)+{0..7},
    // slots 8-15 hold K = 16 + (l<16?0:8)+{0..7}  -> two b128 loads per 32-k block.
    v8f acc = {0.f, 0.f, 0.f, 0.f, 0.f, 0.f, 0.f, 0.f};
    const int bn  = lane & 15;
    const int bko = (lane < 16) ? 0 : 8;
    #pragma unroll
    for (int kb = 0; kb < 8; ++kb) {
      const int kbase = kb * 32 + bko;
      union { v16h v; uint4 u[2]; } bh, bl;
      const _Float16* p0 = Bhi + bn * BROW + kbase;
      bh.u[0] = *(const uint4*)p0;
      bh.u[1] = *(const uint4*)(p0 + 16);
      const _Float16* p1 = Blo + bn * BROW + kbase;
      bl.u[0] = *(const uint4*)p1;
      bl.u[1] = *(const uint4*)(p1 + 16);
      acc = __builtin_amdgcn_wmma_f32_16x16x32_f16(false, Ahi[kb], false, bh.v,
                                                   (short)0, acc, false, false);
      acc = __builtin_amdgcn_wmma_f32_16x16x32_f16(false, Ahi[kb], false, bl.v,
                                                   (short)0, acc, false, false);
      acc = __builtin_amdgcn_wmma_f32_16x16x32_f16(false, Alo[kb], false, bh.v,
                                                   (short)0, acc, false, false);
    }

    // Threshold-gated exact top-16 maintenance (C layout: VGPR r, lane l ->
    // row r + 8*(l/16), col l%16). Lists are wave-private -> race-free.
    #pragma unroll
    for (int r = 0; r < 8; ++r) {
      const int q = qbase + r + ((lane >> 4) << 3);
      const bool pred = acc[r] > minV[q];
      unsigned long long mask = __ballot(pred);
      while (mask) {
        const int srcl = __ffsll(mask) - 1;
        mask &= mask - 1;
        const float s = __shfl(acc[r], srcl);
        if (lane == 0) {
          const int qq = qbase + r + ((srcl >> 4) << 3);
          const int scol = n0 + (srcl & 15);
          float* S = topS + qq * TOPK;
          int mp = 0; float mn = S[0];
          #pragma unroll
          for (int i2 = 1; i2 < TOPK; ++i2) {
            const float v2 = S[i2];
            if (v2 < mn) { mn = v2; mp = i2; }
          }
          if (s > mn) {
            S[mp] = s;
            topC[qq * TOPK + mp] = scol;
            float nm = S[0];
            #pragma unroll
            for (int i2 = 1; i2 < TOPK; ++i2) nm = fminf(nm, S[i2]);
            minV[qq] = nm;
          }
        }
      }
    }
  }

  __syncthreads();
  for (int i = tid; i < NQ * TOPK; i += 512) {
    candS[blockIdx.x * (NQ * TOPK) + i] = topS[i];
    candC[blockIdx.x * (NQ * TOPK) + i] = topC[i];
  }
}

// ---------------------------------------------------------------------------
// Kernel 3: per query, merge 2000 candidates -> sorted top-16, gather feats.
// One wave per query; 16 rounds of shfl-xor argmax (ties -> lower index).
// ---------------------------------------------------------------------------
__global__ __launch_bounds__(32) void rt_merge_gather(
    const float* __restrict__ candS, const int* __restrict__ candC,
    const float* __restrict__ mem, float* __restrict__ outS,
    float* __restrict__ outI, float* __restrict__ outF) {
  const int q = blockIdx.x;
  const int lane = threadIdx.x;
  float sc[CPL];
  int   cl[CPL];
  #pragma unroll
  for (int i = 0; i < CPL; ++i) {
    const int j = lane + 32 * i;
    if (j < NCAND) {
      const int wg = j >> 4, slot = j & 15;
      sc[i] = candS[wg * (NQ * TOPK) + q * TOPK + slot];
      cl[i] = candC[wg * (NQ * TOPK) + q * TOPK + slot];
    } else {
      sc[i] = -__builtin_inff();
      cl[i] = 0x7fffffff;
    }
  }
  for (int kk = 0; kk < TOPK; ++kk) {
    float best = -__builtin_inff();
    int bcol = 0x7fffffff;
    #pragma unroll
    for (int i = 0; i < CPL; ++i) {
      if (sc[i] > best || (sc[i] == best && cl[i] < bcol)) { best = sc[i]; bcol = cl[i]; }
    }
    float rbest = best;
    int rcol = bcol;
    #pragma unroll
    for (int off = 16; off > 0; off >>= 1) {
      const float ob = __shfl_xor(rbest, off);
      const int   oc = __shfl_xor(rcol, off);
      if (ob > rbest || (ob == rbest && oc < rcol)) { rbest = ob; rcol = oc; }
    }
    bool removed = false;   // columns are unique per query -> exact removal
    #pragma unroll
    for (int i = 0; i < CPL; ++i) {
      if (!removed && sc[i] == rbest && cl[i] == rcol) {
        sc[i] = -__builtin_inff();
        removed = true;
      }
    }
    if (lane == 0) {
      outS[q * TOPK + kk] = rbest;
      outI[q * TOPK + kk] = (float)rcol;   // exact for idx < 2^24
    }
    const float4* gsrc = (const float4*)(mem + (size_t)rcol * D);
    float4* gdst = (float4*)(outF + ((size_t)q * TOPK + kk) * D);
    gdst[lane]      = gsrc[lane];
    gdst[lane + 32] = gsrc[lane + 32];
  }
}

// ---------------------------------------------------------------------------
extern "C" void kernel_launch(void* const* d_in, const int* in_sizes, int n_in,
                              void* d_out, int out_size, void* d_ws, size_t ws_size,
                              hipStream_t stream) {
  (void)in_sizes; (void)n_in; (void)out_size; (void)ws_size;
  const float* hidden = (const float*)d_in[0];
  const float* Wp     = (const float*)d_in[1];
  const float* bp     = (const float*)d_in[2];
  const float* mem    = (const float*)d_in[3];
  // d_in[4] = topk (scalar 16), compile-time constant here.

  char* ws = (char*)d_ws;
  _Float16* qhi = (_Float16*)ws;                         // 256*256*2 = 131072 B
  _Float16* qlo = (_Float16*)(ws + 131072);              // 131072 B
  float* candS  = (float*)(ws + 262144);                 // 125*256*16*4 = 2048000 B
  int*   candC  = (int*)(ws + 262144 + 2048000);         // 2048000 B

  float* outS = (float*)d_out;                 // [256,16] scores
  float* outI = outS + NQ * TOPK;              // [256,16] indices (as float)
  float* outF = outI + NQ * TOPK;              // [256,16,256] gathered feats

  rt_encode_q<<<dim3(NQ), dim3(256), 0, stream>>>(hidden, Wp, bp, qhi, qlo);
  rt_score_topk<<<dim3(NWG_A), dim3(512), 0, stream>>>(qhi, qlo, mem, candS, candC);
  rt_merge_gather<<<dim3(NQ), dim3(32), 0, stream>>>(candS, candC, mem, outS, outI, outF);
}
